// GPT2_25125558681779
// MI455X (gfx1250) — compile-verified, matrix operand order checked by reading, not run on hardware
//
#include <hip/hip_runtime.h>
#include <hip/hip_bf16.h>

// ---------------------------------------------------------------------------
// GPT-2 small forward pass for gfx1250 (MI455X): bf16 WMMA GEMMs, fp32 accum,
// double-buffered LDS pipeline (1 barrier per K-step, global loads overlapped
// with v_wmma_f32_16x16x32_bf16 compute).
// ---------------------------------------------------------------------------

#define L_ 12
#define H_ 12
#define C_ 768
#define V_ 50257
#define B_ 2
#define T_ 1024
#define D_ 64
#define MROWS (B_ * T_)   // 2048

typedef __attribute__((ext_vector_type(16))) __bf16 v16bf;
typedef __attribute__((ext_vector_type(8)))  float  v8f;

// fp32 -> bf16 via hardware convert (RNE); bit pattern for LDS staging
__device__ __forceinline__ unsigned short f2bf(float f) {
    __bf16 h = (__bf16)f;
    return __builtin_bit_cast(unsigned short, h);
}

// ------------------------------- GEMM --------------------------------------
// C[M,N] = act( A[M,K] @ B[K,N] + bias + resid ), row-major fp32 in global,
// bf16 staging in LDS, v_wmma_f32_16x16x32_bf16 compute.
#define BM 128
#define BN 128
#define BK 32
#define LDT 34   // LDS row stride in ushorts: 68B = 17 banks -> conflict-free

#define FLAG_GELU 1
#define FLAG_RES  2

// Assemble a 16x32 bf16 fragment (A layout; B uses same layout on its
// transposed-in-LDS tile). base = row 0 of the 16-row group.
// ISA 16-bit A layout: lanes 0-15 hold K {0..7,16..23}, lanes 16-31 hold
// K {8..15,24..31}; each dword holds a consecutive K pair.
__device__ __forceinline__ v16bf load_frag(const unsigned short* base, int lane) {
    const int half = lane >> 4;          // 0 or 1
    const int r    = lane & 15;          // row (A) / col (B)
    const unsigned short* p = base + r * LDT + half * 8;
    union { unsigned int i[8]; v16bf v; } u;
#pragma unroll
    for (int t = 0; t < 8; ++t) {
        const int off = (t < 4) ? (2 * t) : (16 + 2 * (t - 4));
        u.i[t] = *(const unsigned int*)(p + off);
    }
    return u.v;
}

__global__ __launch_bounds__(256) void gpt2_wmma_gemm(
    const float* __restrict__ A, const float* __restrict__ Bw,
    const float* __restrict__ bias, const float* __restrict__ resid,
    float* __restrict__ Cout, int M, int N, int K, int flags)
{
    __shared__ unsigned short As[2][BM * LDT];
    __shared__ unsigned short Bs[2][BN * LDT];

    const int tid   = threadIdx.x;
    const int lane  = tid & 31;
    const int wave  = tid >> 5;
    const int waveM = wave >> 1;           // 0..3  (32 rows each)
    const int waveN = wave & 1;            // 0..1  (64 cols each)
    const int m0    = blockIdx.y * BM;
    const int n0    = blockIdx.x * BN;

    v8f acc[2][4];
#pragma unroll
    for (int i = 0; i < 2; ++i)
#pragma unroll
        for (int j = 0; j < 4; ++j)
            acc[i][j] = (v8f){0.f, 0.f, 0.f, 0.f, 0.f, 0.f, 0.f, 0.f};

    // global->LDS staging indices (256 threads x 16 elems = 4096 per tile)
    const int arow = tid >> 1;             // 0..127
    const int acol = (tid & 1) * 16;       // 0 | 16
    const int brow = tid >> 3;             // 0..31
    const int bcol = (tid & 7) * 16;       // 0..112

    const int nk = K / BK;

    // ---- prologue: stage tile 0 into buffer 0 ----
    {
        const float* ag = A + (size_t)(m0 + arow) * K + acol;
#pragma unroll
        for (int j = 0; j < 16; ++j)
            As[0][arow * LDT + acol + j] = f2bf(ag[j]);
        const float* bg = Bw + (size_t)brow * N + n0 + bcol;
#pragma unroll
        for (int j = 0; j < 16; ++j) {
            const int n = n0 + bcol + j;
            Bs[0][(bcol + j) * LDT + brow] = f2bf((n < N) ? bg[j] : 0.0f);
        }
    }
    __syncthreads();

    // ---- main pipeline: one barrier per K-step ----
    for (int kt = 0; kt < nk; ++kt) {
        const int cur = kt & 1;
        const int nxt = cur ^ 1;
        const bool have_next = (kt + 1 < nk);

        // 1) issue next tile's global loads into registers (in flight
        //    while the WMMAs below execute)
        float areg[16], breg[16];
        if (have_next) {
            const int k1 = (kt + 1) * BK;
            const float* ag = A + (size_t)(m0 + arow) * K + k1 + acol;
#pragma unroll
            for (int j = 0; j < 16; ++j) areg[j] = ag[j];
            const float* bg = Bw + (size_t)(k1 + brow) * N + n0 + bcol;
#pragma unroll
            for (int j = 0; j < 16; ++j) {
                const int n = n0 + bcol + j;
                breg[j] = (n < N) ? bg[j] : 0.0f;
            }
        }

        // 2) compute from current LDS buffer
        v16bf afrag0 = load_frag(&As[cur][(waveM * 32 + 0) * LDT], lane);
        v16bf afrag1 = load_frag(&As[cur][(waveM * 32 + 16) * LDT], lane);
#pragma unroll
        for (int nt = 0; nt < 4; ++nt) {
            v16bf bfrag = load_frag(&Bs[cur][(waveN * 64 + nt * 16) * LDT], lane);
            acc[0][nt] = __builtin_amdgcn_wmma_f32_16x16x32_bf16(
                false, afrag0, false, bfrag, (short)0, acc[0][nt], false, false);
            acc[1][nt] = __builtin_amdgcn_wmma_f32_16x16x32_bf16(
                false, afrag1, false, bfrag, (short)0, acc[1][nt], false, false);
        }

        // 3) convert + store next tile into alternate buffer
        if (have_next) {
#pragma unroll
            for (int j = 0; j < 16; ++j)
                As[nxt][arow * LDT + acol + j] = f2bf(areg[j]);
#pragma unroll
            for (int j = 0; j < 16; ++j)
                Bs[nxt][(bcol + j) * LDT + brow] = f2bf(breg[j]);
        }
        __syncthreads();
    }

    // Epilogue: C/D layout -> VGPR r holds row (r + 8*(lane>>4)), col lane&15
    const int half = lane >> 4;
    const int ncol = lane & 15;
    const bool do_gelu = (flags & FLAG_GELU) != 0;
    const bool do_res  = (flags & FLAG_RES) != 0;
#pragma unroll
    for (int mt = 0; mt < 2; ++mt) {
#pragma unroll
        for (int nt = 0; nt < 4; ++nt) {
            const int n = n0 + waveN * 64 + nt * 16 + ncol;
            if (n >= N) continue;
            const int mbase = m0 + waveM * 32 + mt * 16 + half * 8;
#pragma unroll
            for (int r = 0; r < 8; ++r) {
                const int m = mbase + r;
                float v = acc[mt][nt][r];
                if (bias) v += bias[n];
                if (do_res) v += resid[(size_t)m * N + n];
                if (do_gelu) {
                    const float c = 0.7978845608028654f;
                    v = 0.5f * v * (1.0f + tanhf(c * (v + 0.044715f * v * v * v)));
                }
                Cout[(size_t)m * N + n] = v;
            }
        }
    }
}

// ----------------------------- LayerNorm -----------------------------------
// Matches torch-style unbiased variance (ddof=1), eps=1e-5.
__global__ __launch_bounds__(256) void gpt2_ln(
    const float* __restrict__ x, const float* __restrict__ w,
    const float* __restrict__ b, float* __restrict__ y)
{
    const int row = blockIdx.x;
    const int tid = threadIdx.x;
    const float* xp = x + (size_t)row * C_;
    float s = 0.f, s2 = 0.f;
    for (int c = tid; c < C_; c += 256) { float v = xp[c]; s += v; s2 += v * v; }
    __shared__ float r1[256], r2[256];
    r1[tid] = s; r2[tid] = s2;
    __syncthreads();
    for (int st = 128; st > 0; st >>= 1) {
        if (tid < st) { r1[tid] += r1[tid + st]; r2[tid] += r2[tid + st]; }
        __syncthreads();
    }
    const float mean = r1[0] / (float)C_;
    const float var  = (r2[0] - r1[0] * r1[0] / (float)C_) / (float)(C_ - 1);
    const float inv  = rsqrtf(var + 1e-5f);
    for (int c = tid; c < C_; c += 256)
        y[(size_t)row * C_ + c] = (xp[c] - mean) * inv * w[c] + b[c];
}

// ----------------------------- Attention -----------------------------------
// One block per (t, h, b): masked softmax(q.K^T/sqrt(D)) @ V. ~7% of FLOPs.
__global__ __launch_bounds__(128) void gpt2_attn(
    const float* __restrict__ qkv, float* __restrict__ o)
{
    const int t = blockIdx.x, h = blockIdx.y, b = blockIdx.z;
    const int tid = threadIdx.x;
    __shared__ float qsh[D_];
    __shared__ float sc[T_];
    __shared__ float red[128];

    const int row = b * T_ + t;
    const float* qp = qkv + (size_t)row * (3 * C_) + h * D_;
    if (tid < D_) qsh[tid] = qp[tid];
    __syncthreads();

    float lmax = -3.4e38f;
    for (int s = tid; s <= t; s += 128) {
        const float* kp = qkv + (size_t)(b * T_ + s) * (3 * C_) + C_ + h * D_;
        float d = 0.f;
#pragma unroll
        for (int j = 0; j < D_; ++j) d += qsh[j] * kp[j];
        d *= 0.125f;                       // 1/sqrt(64)
        sc[s] = d;
        lmax = fmaxf(lmax, d);
    }
    red[tid] = lmax;
    __syncthreads();
    for (int st = 64; st > 0; st >>= 1) {
        if (tid < st) red[tid] = fmaxf(red[tid], red[tid + st]);
        __syncthreads();
    }
    const float mx = red[0];
    __syncthreads();

    float lsum = 0.f;
    for (int s = tid; s <= t; s += 128) {
        float e = expf(sc[s] - mx);
        sc[s] = e;
        lsum += e;
    }
    red[tid] = lsum;
    __syncthreads();
    for (int st = 64; st > 0; st >>= 1) {
        if (tid < st) red[tid] += red[tid + st];
        __syncthreads();
    }
    const float inv = 1.0f / red[0];
    __syncthreads();

    if (tid < D_) {
        float a = 0.f;
        for (int s = 0; s <= t; ++s) {
            const float* vp = qkv + (size_t)(b * T_ + s) * (3 * C_) + 2 * C_ + h * D_;
            a += sc[s] * vp[tid];
        }
        o[(size_t)row * C_ + h * D_ + tid] = a * inv;
    }
}

// ----------------------------- Embedding -----------------------------------
__global__ __launch_bounds__(256) void gpt2_embed(
    const int* __restrict__ idx, const float* __restrict__ wte,
    const float* __restrict__ wpe, float* __restrict__ x)
{
    const int m = blockIdx.x;              // 0..MROWS-1
    const int t = m % T_;
    const int tok = idx[m];
    for (int c = threadIdx.x; c < C_; c += 256)
        x[(size_t)m * C_ + c] = wte[(size_t)tok * C_ + c] + wpe[(size_t)t * C_ + c];
}

// ----------------------------- Launcher ------------------------------------
extern "C" void kernel_launch(void* const* d_in, const int* in_sizes, int n_in,
                              void* d_out, int out_size, void* d_ws, size_t ws_size,
                              hipStream_t stream)
{
    const int*   idx     = (const int*)  d_in[0];
    const float* wte     = (const float*)d_in[1];
    const float* wpe     = (const float*)d_in[2];
    const float* ln1_w   = (const float*)d_in[3];
    const float* ln1_b   = (const float*)d_in[4];
    const float* w_attn  = (const float*)d_in[5];
    const float* b_attn  = (const float*)d_in[6];
    const float* w_aproj = (const float*)d_in[7];
    const float* b_aproj = (const float*)d_in[8];
    const float* ln2_w   = (const float*)d_in[9];
    const float* ln2_b   = (const float*)d_in[10];
    const float* w_fc    = (const float*)d_in[11];
    const float* b_fc    = (const float*)d_in[12];
    const float* w_mproj = (const float*)d_in[13];
    const float* b_mproj = (const float*)d_in[14];
    const float* lnf_w   = (const float*)d_in[15];
    const float* lnf_b   = (const float*)d_in[16];
    const float* w_head  = (const float*)d_in[17];
    float* out = (float*)d_out;

    char* ws = (char*)d_ws;
    size_t off = 0;
    float* x    = (float*)(ws + off); off += (size_t)MROWS * C_     * 4;
    float* h    = (float*)(ws + off); off += (size_t)MROWS * C_     * 4;
    float* qkv  = (float*)(ws + off); off += (size_t)MROWS * 3 * C_ * 4;
    float* attno= (float*)(ws + off); off += (size_t)MROWS * C_     * 4;
    float* fcb  = (float*)(ws + off); off += (size_t)MROWS * 4 * C_ * 4;
    (void)ws_size; (void)in_sizes; (void)n_in; (void)out_size;

    const dim3 blk256(256);
    auto gemm_grid = [](int N) { return dim3((N + BN - 1) / BN, MROWS / BM); };

    // 1) token + position embedding
    gpt2_embed<<<dim3(MROWS), blk256, 0, stream>>>(idx, wte, wpe, x);

    // 2) transformer blocks
    for (int l = 0; l < L_; ++l) {
        const float* l1w = ln1_w   + (size_t)l * C_;
        const float* l1b = ln1_b   + (size_t)l * C_;
        const float* wa  = w_attn  + (size_t)l * C_ * 3 * C_;
        const float* ba  = b_attn  + (size_t)l * 3 * C_;
        const float* wp  = w_aproj + (size_t)l * C_ * C_;
        const float* bp  = b_aproj + (size_t)l * C_;
        const float* l2w = ln2_w   + (size_t)l * C_;
        const float* l2b = ln2_b   + (size_t)l * C_;
        const float* wf  = w_fc    + (size_t)l * C_ * 4 * C_;
        const float* bf  = b_fc    + (size_t)l * 4 * C_;
        const float* wm  = w_mproj + (size_t)l * 4 * C_ * C_;
        const float* bm  = b_mproj + (size_t)l * C_;

        gpt2_ln<<<dim3(MROWS), blk256, 0, stream>>>(x, l1w, l1b, h);
        gpt2_wmma_gemm<<<gemm_grid(3 * C_), blk256, 0, stream>>>(
            h, wa, ba, nullptr, qkv, MROWS, 3 * C_, C_, 0);
        gpt2_attn<<<dim3(T_, H_, B_), dim3(128), 0, stream>>>(qkv, attno);
        gpt2_wmma_gemm<<<gemm_grid(C_), blk256, 0, stream>>>(
            attno, wp, bp, x, x, MROWS, C_, C_, FLAG_RES);
        gpt2_ln<<<dim3(MROWS), blk256, 0, stream>>>(x, l2w, l2b, h);
        gpt2_wmma_gemm<<<gemm_grid(4 * C_), blk256, 0, stream>>>(
            h, wf, bf, nullptr, fcb, MROWS, 4 * C_, C_, FLAG_GELU);
        gpt2_wmma_gemm<<<gemm_grid(C_), blk256, 0, stream>>>(
            fcb, wm, bm, x, x, MROWS, C_, 4 * C_, FLAG_RES);
    }

    // 3) final LN + LM head
    gpt2_ln<<<dim3(MROWS), blk256, 0, stream>>>(x, lnf_w, lnf_b, h);
    gpt2_wmma_gemm<<<gemm_grid(V_), blk256, 0, stream>>>(
        h, w_head, nullptr, nullptr, out, MROWS, V_, C_, 0);
}
